// KernelModel_14723147891043
// MI455X (gfx1250) — matrix operand order repaired and checked by reading, not compiled, not run on hardware
//
#include <hip/hip_runtime.h>
#include <hip/hip_bf16.h>

#define N_SAMPLES 8192
#define N_CENTERS 10000
#define X_DIM     1024
#define Y_DIM     128
#define BW_INV    0.1f

#define BM 128
#define BN 128
#define SA  40    // LDS stride (bf16 elems) for 32-wide bf16 tiles (80B, 16B-aligned)
#define SAF 36    // LDS stride (f32 elems) for 32-wide f32 tiles (144B, 16B-aligned)
#define SK  136   // LDS stride for 128-wide kmat tile (272B, 16B-aligned)
#define SW  136   // LDS stride for transposed weight tile
#define CB_TOTAL 79          // ceil(10000/128)
#define SPLIT 4
#define CB_PER_SPLIT 20      // ceil(79/4)

#define F32_TILE_BYTES (BM * SAF * 4)        // 18432
#define F32_BUF_BYTES  (2 * F32_TILE_BYTES)  // A+B per buffer = 36864
#define POOL_BYTES     (2 * F32_BUF_BYTES)   // double buffer = 73728 (aliases Ks+WT)

typedef __attribute__((ext_vector_type(16))) __bf16 v16bf;
typedef __attribute__((ext_vector_type(8)))  __bf16 v8bf;
typedef __attribute__((ext_vector_type(4)))  __bf16 v4bf;
typedef __attribute__((ext_vector_type(8)))  float  v8f;

union FragBF { v16bf v; v8bf h[2]; };

// A-matrix fragment (16x32 bf16): lane<16 -> row=lane, K {0..7,16..23};
// lane>=16 -> row=lane-16, K {8..15,24..31}. Two 16B LDS loads.
__device__ __forceinline__ FragBF frag_A(const __bf16* tile, int rowbase, int stride,
                                         int kbase, int lane) {
  FragBF f;
  const __bf16* p = tile + (rowbase + (lane & 15)) * stride + kbase + ((lane >= 16) ? 8 : 0);
  f.h[0] = *(const v8bf*)(p);
  f.h[1] = *(const v8bf*)(p + 16);
  return f;
}

// B-matrix fragment (32x16 bf16), tile stored [n][k]: lane<16 -> col=lane K {0..7,16..23},
// lane>=16 -> col=lane-16 K {8..15,24..31}. Two 16B LDS loads.
__device__ __forceinline__ FragBF frag_B(const __bf16* tile, int rowbase, int stride,
                                         int kbase, int lane) {
  FragBF f;
  const __bf16* p = tile + (rowbase + (lane & 15)) * stride + kbase + ((lane >= 16) ? 8 : 0);
  f.h[0] = *(const v8bf*)(p);
  f.h[1] = *(const v8bf*)(p + 16);
  return f;
}

__device__ __forceinline__ void cvt4(float4 v, v4bf& hi, v4bf& lo) {
  float f[4] = {v.x, v.y, v.z, v.w};
#pragma unroll
  for (int i = 0; i < 4; ++i) {
    __bf16 h = (__bf16)f[i];
    hi[i] = h;
    lo[i] = (__bf16)(f[i] - (float)h);
  }
}

// Issue async global->LDS copies (raw f32, 16B per lane per issue) for one
// 128x32 A tile and one 128x32 B tile into the given LDS f32 buffers.
__device__ __forceinline__ void async_tile_loads(const float* __restrict__ samples,
                                                 const float* __restrict__ centers,
                                                 int m0, int cblk, int k0,
                                                 uint32_t ldsA, uint32_t ldsB, int tid) {
#pragma unroll
  for (int j = 0; j < 4; ++j) {
    const int q   = j * 256 + tid;   // float4 chunk id, 0..1023
    const int lin = q * 4;
    const int r   = lin >> 5;
    const int c   = lin & 31;
    {
      uint64_t ga = (uint64_t)(uintptr_t)(samples + (size_t)(m0 + r) * X_DIM + k0 + c);
      uint32_t la = ldsA + (uint32_t)(r * SAF + c) * 4u;
      asm volatile("global_load_async_to_lds_b128 %0, %1, off"
                   :: "v"(la), "v"(ga) : "memory");
    }
    int gr = cblk + r;
    if (gr >= N_CENTERS) gr = N_CENTERS - 1;
    {
      uint64_t ga = (uint64_t)(uintptr_t)(centers + (size_t)gr * X_DIM + k0 + c);
      uint32_t la = ldsB + (uint32_t)(r * SAF + c) * 4u;
      asm volatile("global_load_async_to_lds_b128 %0, %1, off"
                   :: "v"(la), "v"(ga) : "memory");
    }
  }
}

__global__ __launch_bounds__(256)
void laplace_norms_kernel(const float* __restrict__ samples,
                          const float* __restrict__ centers,
                          float* __restrict__ xsq, float* __restrict__ zsq) {
  __shared__ float red[256];
  const int row = blockIdx.x;
  const float* p = (row < N_SAMPLES) ? samples + (size_t)row * X_DIM
                                     : centers + (size_t)(row - N_SAMPLES) * X_DIM;
  float s = 0.f;
  for (int i = threadIdx.x; i < X_DIM; i += 256) { float v = p[i]; s += v * v; }
  red[threadIdx.x] = s;
  __syncthreads();
#pragma unroll
  for (int st = 128; st > 0; st >>= 1) {
    if ((int)threadIdx.x < st) red[threadIdx.x] += red[threadIdx.x + st];
    __syncthreads();
  }
  if (threadIdx.x == 0) {
    if (row < N_SAMPLES) xsq[row] = red[0];
    else                 zsq[row - N_SAMPLES] = red[0];
  }
}

__global__ __launch_bounds__(256)
void laplace_zero_kernel(float* __restrict__ out) {
  out[(size_t)blockIdx.x * 256 + threadIdx.x] = 0.0f;
}

__global__ __launch_bounds__(256)
void laplace_fused_kernel(const float* __restrict__ samples,
                          const float* __restrict__ centers,
                          const float* __restrict__ weight,
                          const float* __restrict__ xsq,
                          const float* __restrict__ zsq,
                          float* __restrict__ out) {
  // Aliased pool: during the K-loop it holds double-buffered f32 A/B tiles
  // (filled by async loads); after the K-loop it holds Ks (kmat, bf16) + WT.
  __shared__ __align__(16) char pool[POOL_BYTES];
  __shared__ __bf16 As_hi[BM * SA], As_lo[BM * SA];
  __shared__ __bf16 Bs_hi[BN * SA], Bs_lo[BN * SA];
  __shared__ float  xs[BM], zs[BN];

  const uint32_t pool_lds = (uint32_t)(uintptr_t)(&pool[0]);
  float* fbuf[2][2];
  fbuf[0][0] = (float*)(pool);                                  // buf0 A
  fbuf[0][1] = (float*)(pool + F32_TILE_BYTES);                 // buf0 B
  fbuf[1][0] = (float*)(pool + F32_BUF_BYTES);                  // buf1 A
  fbuf[1][1] = (float*)(pool + F32_BUF_BYTES + F32_TILE_BYTES); // buf1 B
  __bf16* Ks = (__bf16*)(pool);                   // kmat tile [m][c] (after K-loop)
  __bf16* WT = (__bf16*)(pool + F32_BUF_BYTES);   // weight transposed [y][c]

  const int tid  = threadIdx.x;
  const int lane = tid & 31;
  const int w    = tid >> 5;   // wave 0..7
  const int wm   = w & 3;      // 4 waves along M (32 rows each)
  const int wn   = w >> 2;     // 2 waves along N (64 cols each)
  const int m0   = blockIdx.x * BM;

  const int mrow_off = (lane >= 16) ? 8 : 0;
  const int ncol     = lane & 15;

  if (tid < BM) xs[tid] = xsq[m0 + tid];

  const v8f vzero = {};
  v8f pred[8];
#pragma unroll
  for (int i = 0; i < 8; ++i) pred[i] = vzero;

  const int cb0 = blockIdx.y * CB_PER_SPLIT;
  const int cb1 = (cb0 + CB_PER_SPLIT < CB_TOTAL) ? cb0 + CB_PER_SPLIT : CB_TOTAL;

  for (int cb = cb0; cb < cb1; ++cb) {
    const int cblk = cb * BN;
    // All waves are past phase-2 LDS reads of the previous iteration before
    // the async engine may overwrite the aliased pool.
    __syncthreads();
    if (tid < BN) {
      int c = cblk + tid;
      zs[tid] = (c < N_CENTERS) ? zsq[c] : 3.0e28f;  // OOB centers -> k = 0
    }

    v8f acc[2][4];
#pragma unroll
    for (int mt = 0; mt < 2; ++mt)
#pragma unroll
      for (int nt = 0; nt < 4; ++nt) acc[mt][nt] = vzero;

    // Prime the async pipeline: k-step 0 into buffer 0.
    async_tile_loads(samples, centers, m0, cblk, 0,
                     pool_lds, pool_lds + F32_TILE_BYTES, tid);

    // ---- phase 1: 128x128 dot tile via bf16 hi/lo split over K=1024 ----
    for (int ks = 0; ks < X_DIM / 32; ++ks) {
      const int cur = ks & 1;
      // Data for step ks has landed in fbuf[cur] once every wave's ASYNCcnt
      // drained and the barrier completed.
      asm volatile("s_wait_asynccnt 0x0" ::: "memory");
      __syncthreads();

      if (ks + 1 < X_DIM / 32) {
        const uint32_t nb = pool_lds + (uint32_t)((ks + 1) & 1) * F32_BUF_BYTES;
        async_tile_loads(samples, centers, m0, cblk, (ks + 1) * 32,
                         nb, nb + F32_TILE_BYTES, tid);
      }

      // Stage: LDS f32 -> LDS bf16 hi/lo.
      const float* fA = fbuf[cur][0];
      const float* fB = fbuf[cur][1];
#pragma unroll
      for (int j = 0; j < 4; ++j) {
        const int q   = j * 256 + tid;
        const int lin = q * 4;
        const int r   = lin >> 5;
        const int c   = lin & 31;
        v4bf hi, lo;
        float4 a4 = *(const float4*)(fA + r * SAF + c);
        cvt4(a4, hi, lo);
        *(v4bf*)&As_hi[r * SA + c] = hi;
        *(v4bf*)&As_lo[r * SA + c] = lo;
        float4 b4 = *(const float4*)(fB + r * SAF + c);
        cvt4(b4, hi, lo);
        *(v4bf*)&Bs_hi[r * SA + c] = hi;
        *(v4bf*)&Bs_lo[r * SA + c] = lo;
      }
      __syncthreads();

      FragBF ahi[2], alo[2], bhi[4], blo[4];
#pragma unroll
      for (int mt = 0; mt < 2; ++mt) {
        ahi[mt] = frag_A(As_hi, wm * 32 + mt * 16, SA, 0, lane);
        alo[mt] = frag_A(As_lo, wm * 32 + mt * 16, SA, 0, lane);
      }
#pragma unroll
      for (int nt = 0; nt < 4; ++nt) {
        bhi[nt] = frag_B(Bs_hi, wn * 64 + nt * 16, SA, 0, lane);
        blo[nt] = frag_B(Bs_lo, wn * 64 + nt * 16, SA, 0, lane);
      }
#pragma unroll
      for (int mt = 0; mt < 2; ++mt)
#pragma unroll
        for (int nt = 0; nt < 4; ++nt) {
          acc[mt][nt] = __builtin_amdgcn_wmma_f32_16x16x32_bf16(
              false, ahi[mt].v, false, bhi[nt].v, (short)0, acc[mt][nt], false, false);
          acc[mt][nt] = __builtin_amdgcn_wmma_f32_16x16x32_bf16(
              false, ahi[mt].v, false, blo[nt].v, (short)0, acc[mt][nt], false, false);
          acc[mt][nt] = __builtin_amdgcn_wmma_f32_16x16x32_bf16(
              false, alo[mt].v, false, bhi[nt].v, (short)0, acc[mt][nt], false, false);
        }
      // Next iteration's top barrier protects the bf16 staging tiles.
    }

    // ---- epilogue: kmat = exp(-sqrt(max(x2+z2-2*dot, eps))/bw) -> LDS bf16 ----
    // (ASYNCcnt is 0 here; all f32-pool reads retired at the last stage barrier,
    //  so writing the aliased Ks/WT region is safe.)
#pragma unroll
    for (int mt = 0; mt < 2; ++mt)
#pragma unroll
      for (int nt = 0; nt < 4; ++nt)
#pragma unroll
        for (int r = 0; r < 8; ++r) {
          const int ml = wm * 32 + mt * 16 + r + mrow_off;
          const int cl = wn * 64 + nt * 16 + ncol;
          float dot = acc[mt][nt][r];
          float d2  = xs[ml] + zs[cl] - 2.0f * dot;
          float d   = __builtin_sqrtf(fmaxf(d2, 1e-12f));
          Ks[ml * SK + cl] = (__bf16)__expf(-d * BW_INV);
        }

    // ---- load weight tile transposed into LDS as bf16 ----
#pragma unroll
    for (int j = 0; j < 16; ++j) {
      const int q   = j * 256 + tid;  // float4 chunk id over 128x128 elems
      const int lin = q * 4;
      const int y   = lin & 127;
      const int c   = lin >> 7;
      int gc = cblk + c;
      if (gc >= N_CENTERS) gc = N_CENTERS - 1;
      float4 w4 = *(const float4*)(weight + (size_t)gc * Y_DIM + y);
      WT[(y + 0) * SW + c] = (__bf16)w4.x;
      WT[(y + 1) * SW + c] = (__bf16)w4.y;
      WT[(y + 2) * SW + c] = (__bf16)w4.z;
      WT[(y + 3) * SW + c] = (__bf16)w4.w;
    }
    __syncthreads();

    // ---- phase 2: pred += kmat_tile @ weight_tile (bf16 WMMA) ----
#pragma unroll
    for (int kc = 0; kc < 4; ++kc) {
      FragBF ka = frag_A(Ks, w * 16, SK, kc * 32, lane);
#pragma unroll
      for (int nt = 0; nt < 8; ++nt) {
        FragBF wb = frag_B(WT, nt * 16, SW, kc * 32, lane);
        pred[nt] = __builtin_amdgcn_wmma_f32_16x16x32_bf16(
            false, ka.v, false, wb.v, (short)0, pred[nt], false, false);
      }
    }
  }

  // ---- accumulate into global output (centers split across blockIdx.y) ----
#pragma unroll
  for (int nt = 0; nt < 8; ++nt)
#pragma unroll
    for (int r = 0; r < 8; ++r) {
      const int row = m0 + w * 16 + r + mrow_off;
      const int y   = nt * 16 + ncol;
      atomicAdd(&out[(size_t)row * Y_DIM + y], pred[nt][r]);
    }
}

extern "C" void kernel_launch(void* const* d_in, const int* in_sizes, int n_in,
                              void* d_out, int out_size, void* d_ws, size_t ws_size,
                              hipStream_t stream) {
  const float* samples = (const float*)d_in[0];
  const float* centers = (const float*)d_in[1];
  const float* weight  = (const float*)d_in[2];
  float* out = (float*)d_out;
  float* xsq = (float*)d_ws;            // [8192]
  float* zsq = xsq + N_SAMPLES;         // [10000]

  laplace_zero_kernel<<<(N_SAMPLES * Y_DIM) / 256, 256, 0, stream>>>(out);
  laplace_norms_kernel<<<N_SAMPLES + N_CENTERS, 256, 0, stream>>>(samples, centers, xsq, zsq);

  dim3 grid(N_SAMPLES / BM, SPLIT);
  laplace_fused_kernel<<<grid, 256, 0, stream>>>(samples, centers, weight, xsq, zsq, out);
}
